// DualOutputMoE_22634477650571
// MI455X (gfx1250) — matrix operand order repaired and checked by reading, not compiled, not run on hardware
//
#include <hip/hip_runtime.h>

// Problem constants (from reference): B=4, S=1024, D=1024, F=4096, E=8, TOP_K=2
#define T_TOK 4096   // B*S tokens
#define DIM   1024   // D
#define FF    4096   // F
#define NE    8      // experts
#define TM    32     // tokens per FFN tile (one workgroup)
#define FC    128    // F-chunk held in LDS between the two GEMMs
#define KC    64     // K-slab depth staged by the Tensor Data Mover

typedef float  v8f   __attribute__((ext_vector_type(8)));
typedef __bf16 v16bf __attribute__((ext_vector_type(16)));
typedef __bf16 v8bf  __attribute__((ext_vector_type(8)));
typedef __bf16 v4bf  __attribute__((ext_vector_type(4)));
typedef unsigned int v4ui __attribute__((ext_vector_type(4)));
typedef int    v8i   __attribute__((ext_vector_type(8)));
typedef int    v4i   __attribute__((ext_vector_type(4)));

// ---------------------------------------------------------------------------
// TDM descriptor (CDNA5 ISA ch.8): 2-D tile load global->LDS, bf16 elements.
// Per-wave tile: tile_dim0 = KC elems (row, contiguous), tile_dim1 = 8 rows,
// row stride = DIM elems. LDS padding: after every 32 DWORDs (=128B row) add
// 4 DWORDs (16B) -> LDS pitch 144B, matching lds_w1[..][KC+8] and giving
// conflict-free ds_load_b128 fragment reads.
//   group0: [1:0]=count=1 | lds_addr | global_addr[56:0] | type=2
//   group1: data_size=1(2B), pad_en, pad_interval=4(32DW), pad_amount=3(4DW),
//           tensor_dim0=2^30, tensor_dim1=2^20 (tiles always interior),
//           tile_dim0=KC, tile_dim1=8, stride0=DIM
// This toolchain exposes the 6-arg builtin:
//   (uint32x4 g0, int32x8 g1, int32x4 g2, int32x4 g3, int32x8 extra, i32 cpol)
// groups 2/3 + extra are zero for a 2-D tensor.
// ---------------------------------------------------------------------------
__device__ __forceinline__ void tdm_issue_w1(const __bf16* gsrc,
                                             unsigned lds_byte_addr) {
  const unsigned long long ga = (unsigned long long)(size_t)gsrc;
  v4ui g0;
  g0[0] = 1u;                                   // count=1, user mode
  g0[1] = lds_byte_addr;                        // LDS dest (bytes)
  g0[2] = (unsigned)ga;                         // global_addr[31:0]
  g0[3] = ((unsigned)(ga >> 32) & 0x1ffffffu) | 0x80000000u;  // [56:32]|type=2
  v8i g1;
  g1[0] = (int)0x07110000u;   // data_size=1 | pad_en | interval=4 | amount=3
  g1[1] = 0;                  // atomic_barrier=0, tensor_dim0[15:0]=0
  g1[2] = 0x4000;             // tensor_dim0[31:16]=0x4000 (dim0 = 2^30)
  g1[3] = 0x10 | (KC << 16);  // tensor_dim1[31:16]=0x10 (2^20), tile_dim0=KC
  g1[4] = 8;                  // tile_dim1 = 8 rows, tile_dim2 = 0
  g1[5] = DIM;                // tensor_dim0_stride[31:0]
  g1[6] = 0;                  // stride0[47:32]=0, stride1[15:0]=0
  g1[7] = 0;
  const v4i z4 = {0, 0, 0, 0};                  // groups 2/3 unused (2-D)
  const v8i z8 = {0, 0, 0, 0, 0, 0, 0, 0};
  __builtin_amdgcn_tensor_load_to_lds(g0, g1, z4, z4, z8, 0);
}

// ---------------------------------------------------------------------------
// Pre-pass 1: x fp32 -> bf16 (straight copy-convert, coalesced).
// ---------------------------------------------------------------------------
__global__ __launch_bounds__(256) void cvt_x_kernel(
    const float* __restrict__ x, __bf16* __restrict__ xb)
{
  const size_t i = (size_t)(blockIdx.x * blockDim.x + threadIdx.x) * 4;
  const float4 v = *(const float4*)(x + i);
  v4bf o;
  o[0] = (__bf16)v.x; o[1] = (__bf16)v.y; o[2] = (__bf16)v.z; o[3] = (__bf16)v.w;
  *(v4bf*)(xb + i) = o;
}

// ---------------------------------------------------------------------------
// Pre-pass 2: per-expert transpose + fp32->bf16. src[e][R][C] -> dst[e][C][R].
// 32x32 tiles via LDS, coalesced on both sides. grid = (C/32, R/32, E).
// ---------------------------------------------------------------------------
__global__ __launch_bounds__(256) void transpose_bf16_kernel(
    const float* __restrict__ src, __bf16* __restrict__ dst, int R, int C)
{
  __shared__ float tile[32][33];
  const int e  = blockIdx.z;
  const int c0 = blockIdx.x * 32, r0 = blockIdx.y * 32;
  const float* s = src + (size_t)e * R * C;
  __bf16*      d = dst + (size_t)e * R * C;
  const int tx = threadIdx.x & 31, ty = threadIdx.x >> 5;   // 32 x 8
  #pragma unroll
  for (int i = 0; i < 32; i += 8)
    tile[ty + i][tx] = s[(size_t)(r0 + ty + i) * C + (c0 + tx)];
  __syncthreads();
  #pragma unroll
  for (int i = 0; i < 32; i += 8)
    d[(size_t)(c0 + ty + i) * R + (r0 + tx)] = (__bf16)tile[tx][ty + i];
}

// ---------------------------------------------------------------------------
// Gating: one wave32 per token; butterfly-reduce 8 logits; softmax + top-2.
// ---------------------------------------------------------------------------
__global__ __launch_bounds__(256) void gate_topk_kernel(
    const float* __restrict__ x, const float* __restrict__ gw,
    const float* __restrict__ gb, float* __restrict__ cw8)
{
  const int lane = threadIdx.x & 31;
  const int wave = threadIdx.x >> 5;
  const int t    = blockIdx.x * 8 + wave;

  float acc[NE];
  #pragma unroll
  for (int e = 0; e < NE; ++e) acc[e] = 0.f;

  const float* xr = x + (size_t)t * DIM;
  for (int d = lane; d < DIM; d += 32) {
    const float  xv = xr[d];
    const float4 w0 = ((const float4*)(gw + (size_t)d * NE))[0];
    const float4 w1 = ((const float4*)(gw + (size_t)d * NE))[1];
    acc[0] += xv * w0.x; acc[1] += xv * w0.y;
    acc[2] += xv * w0.z; acc[3] += xv * w0.w;
    acc[4] += xv * w1.x; acc[5] += xv * w1.y;
    acc[6] += xv * w1.z; acc[7] += xv * w1.w;
  }
  #pragma unroll
  for (int m = 16; m >= 1; m >>= 1) {
    #pragma unroll
    for (int e = 0; e < NE; ++e) acc[e] += __shfl_xor(acc[e], m, 32);
  }
  if (lane == 0) {
    float lg[NE], p[NE];
    float mx = -1e30f;
    #pragma unroll
    for (int e = 0; e < NE; ++e) { lg[e] = acc[e] + gb[e]; mx = fmaxf(mx, lg[e]); }
    float s = 0.f;
    #pragma unroll
    for (int e = 0; e < NE; ++e) { p[e] = __expf(lg[e] - mx); s += p[e]; }
    int i1 = 0;
    #pragma unroll
    for (int e = 1; e < NE; ++e) if (p[e] > p[i1]) i1 = e;
    int i2 = (i1 == 0) ? 1 : 0;
    #pragma unroll
    for (int e = 0; e < NE; ++e) if (e != i1 && e != i2 && p[e] > p[i2]) i2 = e;
    const float inv = 1.f / s;
    #pragma unroll
    for (int e = 0; e < NE; ++e)
      cw8[(size_t)t * NE + e] = (e == i1) ? p[i1] * inv
                              : (e == i2) ? p[i2] * inv : 0.f;
  }
}

// ---------------------------------------------------------------------------
// Bucket tokens per expert (atomic append; each token lands in 2 lists).
// ---------------------------------------------------------------------------
__global__ __launch_bounds__(256) void bucket_kernel(
    const float* __restrict__ cw8, int* __restrict__ counts,
    int* __restrict__ list)
{
  const int t = blockIdx.x * blockDim.x + threadIdx.x;
  if (t >= T_TOK) return;
  #pragma unroll
  for (int e = 0; e < NE; ++e) {
    if (cw8[(size_t)t * NE + e] > 0.f) {
      const int pos = atomicAdd(&counts[e], 1);
      list[e * T_TOK + pos] = t;
    }
  }
}

// ---------------------------------------------------------------------------
// Grouped expert FFN, bf16 WMMA. 512 threads = 16 waves (2 M x 8 N).
// GEMM1: h(32x128) = x(32x1024) @ W1 chunk; B slabs DMA'd by TDM
//        (double-buffered, s_wait_tensorcnt), A from LDS x-tile.
// GEMM2 (transposed): y^T(1024x32) += W2^T rows (global bf16, K-contig)
//        x h^T (LDS). All fragments are 2x16B vector loads, no converts.
// WMMA layouts per ISA 7.12.2 (wave32, 16x16x32 bf16):
//   A: lane<16 M=lane, elem i K=(i<8?i:i+8); lane>=16 same M, K+8.
//   B: lane<16 N=lane, elem i K=(i<8?i:i+8); lane>=16 K+8 (same as A).
//   C/D: reg r -> M=r+8*(lane/16), N=lane%16.
// ---------------------------------------------------------------------------
__global__ __launch_bounds__(512) void moe_ffn_kernel(
    const __bf16* __restrict__ xb,  const __bf16* __restrict__ w1t,
    const float* __restrict__ b1,   const __bf16* __restrict__ w2t,
    const float* __restrict__ b2,   const float* __restrict__ cw8,
    const int* __restrict__ counts, const int* __restrict__ list,
    float* __restrict__ out)
{
  const int e    = blockIdx.x;
  const int cnt  = counts[e];
  const int row0 = blockIdx.y * TM;
  if (row0 >= cnt) return;                        // uniform block exit

  __shared__ __bf16 lds_x[TM][DIM + 8];           // 66,048 B  (pitch 2064)
  __shared__ __bf16 lds_w1[2][FC][KC + 8];        // 36,864 B  (pitch 144 = TDM pad)
  __shared__ __bf16 lds_h[TM][FC + 8];            //  8,704 B  (pitch 272)
  __shared__ int    tok_s[TM];
  __shared__ float  p_s[TM];

  const int tid = threadIdx.x;
  if (tid < TM) {
    const int r  = row0 + tid;
    const int tk = (r < cnt) ? list[e * T_TOK + r] : 0;
    tok_s[tid] = tk;
    p_s[tid]   = (r < cnt) ? cw8[(size_t)tk * NE + e] : 0.f;
  }
  __syncthreads();

  // Stage the 32-token x tile (bf16) into LDS, coalesced 16B copies.
  {
    const int row = tid >> 4, sub = tid & 15;
    const __bf16* src = xb + (size_t)tok_s[row] * DIM + sub * 64;
    __bf16*       dst = &lds_x[row][sub * 64];
    #pragma unroll
    for (int i = 0; i < 8; ++i)
      ((v8bf*)dst)[i] = ((const v8bf*)src)[i];
  }
  __syncthreads();

  const int wave = tid >> 5, lane = tid & 31;
  const int wm = wave >> 3, wn = wave & 7;        // 2 x 8 wave grid
  const int hl = lane >> 4;                       // lane half
  const int ln = lane & 15;

  const __bf16* w1te = w1t + (size_t)e * FF * DIM;   // [F][D], K-contiguous rows
  const __bf16* w2te = w2t + (size_t)e * DIM * FF;   // [D][F], K-contiguous rows

  v8f acc[8];                                     // y^T: n = 128*wn + 16*nt + .., m-group wm
  #pragma unroll
  for (int nt = 0; nt < 8; ++nt)
    #pragma unroll
    for (int i = 0; i < 8; ++i) acc[nt][i] = 0.f;

  for (int f0 = 0; f0 < FF; f0 += FC) {
    // ================= GEMM 1: h tile (wm, wn) ============================
    v8f hc = {0.f, 0.f, 0.f, 0.f, 0.f, 0.f, 0.f, 0.f};
    // Each wave DMAs its own 8 rows of the slab (rows f0+8*wave..+7).
    const __bf16* wsrc = w1te + (size_t)(f0 + wave * 8) * DIM;
    tdm_issue_w1(wsrc, (unsigned)(size_t)&lds_w1[0][wave * 8][0]);
    for (int ks0 = 0; ks0 < DIM; ks0 += KC) {
      const int  buf  = (ks0 / KC) & 1;
      const bool more = (ks0 + KC) < DIM;
      if (more) {
        tdm_issue_w1(wsrc + (ks0 + KC),
                     (unsigned)(size_t)&lds_w1[buf ^ 1][wave * 8][0]);
        __builtin_amdgcn_s_wait_tensorcnt((short)1);   // current slab arrived
      } else {
        __builtin_amdgcn_s_wait_tensorcnt((short)0);
      }
      __syncthreads();                                 // all waves' slices in
      #pragma unroll
      for (int kk = 0; kk < KC; kk += 32) {
        // A from LDS x tile (rows = tokens 16*wm..)
        const __bf16* ap = &lds_x[16 * wm + ln][(ks0 + kk) + 8 * hl];
        const v8bf alo = *(const v8bf*)ap;
        const v8bf ahi = *(const v8bf*)(ap + 16);
        v16bf a;
        #pragma unroll
        for (int i = 0; i < 8; ++i) { a[i] = alo[i]; a[8 + i] = ahi[i]; }
        // B from TDM slab (row = f-col 16*wn+ln, K contiguous)
        const __bf16* bp = &lds_w1[buf][16 * wn + ln][kk + 8 * hl];
        const v8bf blo = *(const v8bf*)bp;
        const v8bf bhi = *(const v8bf*)(bp + 16);
        v16bf b;
        #pragma unroll
        for (int i = 0; i < 8; ++i) { b[i] = blo[i]; b[8 + i] = bhi[i]; }
        hc = __builtin_amdgcn_wmma_f32_16x16x32_bf16(false, a, false, b,
                                                     (short)0, hc, false, false);
      }
      __syncthreads();                 // done reading buf before it's re-DMA'd
    }
    // h epilogue: +b1, relu, bf16 -> LDS
    const float bias = b1[e * FF + f0 + 16 * wn + ln];
    #pragma unroll
    for (int r = 0; r < 8; ++r) {
      float v = hc[r] + bias;
      v = v > 0.f ? v : 0.f;
      lds_h[16 * wm + r + 8 * hl][16 * wn + ln] = (__bf16)v;
    }
    __syncthreads();

    // ================= GEMM 2 (transposed): y^T += W2^T . h^T =============
    #pragma unroll
    for (int kk = 0; kk < FC; kk += 32) {
      // B fragment = h^T: lane column = token 16*wm+ln, K contiguous in lds_h
      const __bf16* hb = &lds_h[16 * wm + ln][kk + 8 * hl];
      const v8bf blo = *(const v8bf*)hb;
      const v8bf bhi = *(const v8bf*)(hb + 16);
      v16bf bh;
      #pragma unroll
      for (int i = 0; i < 8; ++i) { bh[i] = blo[i]; bh[8 + i] = bhi[i]; }
      #pragma unroll
      for (int nt = 0; nt < 8; ++nt) {
        // A fragment = W2^T rows n = 128*wn+16*nt+ln, K contiguous (global bf16)
        const __bf16* ag = w2te + (size_t)(128 * wn + 16 * nt + ln) * FF
                                + (f0 + kk + 8 * hl);
        const v8bf alo = *(const v8bf*)ag;
        const v8bf ahi = *(const v8bf*)(ag + 16);
        v16bf aw;
        #pragma unroll
        for (int i = 0; i < 8; ++i) { aw[i] = alo[i]; aw[8 + i] = ahi[i]; }
        acc[nt] = __builtin_amdgcn_wmma_f32_16x16x32_bf16(false, aw, false, bh,
                                                          (short)0, acc[nt],
                                                          false, false);
      }
    }
    __syncthreads();                               // before lds_h is rewritten
  }

  // ---- epilogue: +b2, scale by routing prob, atomic accumulate ------------
  const int   m   = 16 * wm + ln;                  // this lane's token slot
  const int   tok = tok_s[m];
  const float p   = p_s[m];
  const float* b2e = b2 + e * DIM;
  #pragma unroll
  for (int nt = 0; nt < 8; ++nt) {
    const int nb = 128 * wn + 16 * nt;
    #pragma unroll
    for (int r = 0; r < 8; ++r) {
      const int   col = nb + r + 8 * hl;
      const float val = (acc[nt][r] + b2e[col]) * p;
      __hip_atomic_fetch_add(out + (size_t)tok * DIM + col, val,
                             __ATOMIC_RELAXED, __HIP_MEMORY_SCOPE_AGENT);
    }
  }
}

// ---------------------------------------------------------------------------
// Workspace layout (requires ~143 MB):
//   [xb  : T*D bf16          8,388,608 B]
//   [w1t : E*F*D bf16       67,108,864 B]
//   [w2t : E*D*F bf16       67,108,864 B]
//   [cw8 : T*8 f32             131,072 B]
//   [counts: 8 i32 (pad 256)]
//   [list: 8*T i32             131,072 B]
// ---------------------------------------------------------------------------
extern "C" void kernel_launch(void* const* d_in, const int* in_sizes, int n_in,
                              void* d_out, int out_size, void* d_ws, size_t ws_size,
                              hipStream_t stream) {
  const float* x  = (const float*)d_in[0];
  const float* gw = (const float*)d_in[1];
  const float* gb = (const float*)d_in[2];
  const float* w1 = (const float*)d_in[3];
  const float* b1 = (const float*)d_in[4];
  const float* w2 = (const float*)d_in[5];
  const float* b2 = (const float*)d_in[6];
  float* out = (float*)d_out;

  char* ws = (char*)d_ws;
  const size_t xb_bytes  = (size_t)T_TOK * DIM * 2;        //  8,388,608
  const size_t w_bytes   = (size_t)NE * DIM * FF * 2;      // 67,108,864
  __bf16* xb   = (__bf16*)ws;
  __bf16* w1t  = (__bf16*)(ws + xb_bytes);
  __bf16* w2t  = (__bf16*)(ws + xb_bytes + w_bytes);
  float*  cw8  = (float*) (ws + xb_bytes + 2 * w_bytes);
  int*    counts = (int*) (ws + xb_bytes + 2 * w_bytes + (size_t)T_TOK * NE * 4);
  int*    list   = (int*) ((char*)counts + 256);

  (void)hipMemsetAsync(out, 0, (size_t)T_TOK * DIM * sizeof(float), stream);
  (void)hipMemsetAsync(counts, 0, NE * sizeof(int), stream);

  // Pre-convert/transpose to bf16, K-contiguous layouts.
  cvt_x_kernel<<<dim3((T_TOK * DIM / 4) / 256), dim3(256), 0, stream>>>(x, xb);
  transpose_bf16_kernel<<<dim3(FF / 32, DIM / 32, NE), dim3(256), 0, stream>>>(
      w1, w1t, DIM, FF);   // w1[e][D][F] -> w1t[e][F][D]
  transpose_bf16_kernel<<<dim3(DIM / 32, FF / 32, NE), dim3(256), 0, stream>>>(
      w2, w2t, FF, DIM);   // w2[e][F][D] -> w2t[e][D][F]

  gate_topk_kernel<<<dim3(T_TOK / 8), dim3(256), 0, stream>>>(x, gw, gb, cw8);
  bucket_kernel<<<dim3(T_TOK / 256), dim3(256), 0, stream>>>(cw8, counts, list);

  moe_ffn_kernel<<<dim3(NE, T_TOK / TM), dim3(512), 0, stream>>>(
      xb, w1t, b1, w2t, b2, cw8, counts, list, out);
}
